// RecurrentNet_4733053960301
// MI455X (gfx1250) — matrix operand order repaired and checked by reading, not compiled, not run on hardware
//
#include <hip/hip_runtime.h>

typedef __attribute__((ext_vector_type(16))) _Float16 v16h;
typedef __attribute__((ext_vector_type(8)))  _Float16 v8h;
typedef __attribute__((ext_vector_type(2)))  __fp16   v2hf;   // cvt_pkrtz result type
typedef __attribute__((ext_vector_type(8)))  float    v8f;
typedef __attribute__((ext_vector_type(4)))  float    v4f;

#define ALPHA  0.2f
#define N_IN   32
#define N_HID  256
#define N_OUT  8
#define BB     64
#define TT     2048
#define KPAD   296                     // padded f16 row stride: 592B, 16B aligned, bank-friendly
#define WCAT_BYTES ((size_t)N_HID * KPAD * 2)       // 151552
#define ABUF_BYTES ((size_t)2 * BB * KPAD * 2)      // 75776 (double-buffered A)
#define XSTG_BYTES ((size_t)2 * BB * N_IN * 4)      // 16384 (double-buffered x_t f32)

#define USE_ASYNC_LDS 1   // global_load_async_to_lds_b32 + s_wait_asynccnt pipeline

__device__ __forceinline__ v16h load_frag16(const _Float16* p0, const _Float16* p1) {
    v8h lo = *(const v8h*)p0;
    v8h hi = *(const v8h*)p1;
    return __builtin_shufflevector(lo, hi, 0,1,2,3,4,5,6,7,8,9,10,11,12,13,14,15);
}

// ---------------------------------------------------------------------------
// Kernel 0: fold time-invariant linears into the recurrence.
//   Wcomb = Wih @ in_W  (256x32),  bcat = bhh + bih + Wih @ in_b
// ---------------------------------------------------------------------------
__global__ void prep_kernel(const float* __restrict__ in_W, const float* __restrict__ in_b,
                            const float* __restrict__ Wih,  const float* __restrict__ bih,
                            const float* __restrict__ bhh,
                            float* __restrict__ Wcomb, float* __restrict__ bcat) {
    const int g = threadIdx.x;            // 256 threads, 1 block
    float bb = bhh[g] + bih[g];
    for (int h = 0; h < N_HID; ++h) bb += Wih[g * N_HID + h] * in_b[h];
    bcat[g] = bb;
    for (int i = 0; i < N_IN; ++i) {
        float acc = 0.f;
        for (int h = 0; h < N_HID; ++h) acc += Wih[g * N_HID + h] * in_W[h * N_IN + i];
        Wcomb[g * N_IN + i] = acc;
    }
}

// ---------------------------------------------------------------------------
// Kernel 1: sequential scan, one persistent workgroup (32 wave32) on one WGP.
//  - B fragments ([Whh|Wcomb] f16) hoisted into registers, loaded once
//  - h state lives in registers (16 f32/lane) for all 2048 steps
//  - A buffer double-buffered in LDS -> exactly ONE barrier per step
//  - x_{t+2} prefetched via GLOBAL_LOAD_ASYNC_TO_LDS (ASYNCcnt), 2-deep pipe
// ---------------------------------------------------------------------------
__launch_bounds__(1024, 1)
__global__ void scan_kernel(const float* __restrict__ input,  // [B,T,I]
                            const float* __restrict__ Whh,    // [H,H]
                            const float* __restrict__ Wcomb,  // [H,I]
                            const float* __restrict__ bcat,   // [H]
                            float* __restrict__ hidden)       // [B,T,H]
{
    extern __shared__ char smem[];
    _Float16* wcat  = (_Float16*)smem;                           // [H][KPAD]
    _Float16* abuf0 = (_Float16*)(smem + WCAT_BYTES);            // [B][KPAD]
    _Float16* abuf1 = abuf0 + (size_t)BB * KPAD;                 // [B][KPAD]
    float*    xstg  = (float*)(smem + WCAT_BYTES + ABUF_BYTES);  // [2][B*N_IN]

    const int tid   = threadIdx.x;
    const int lane  = tid & 31;
    const int wave  = tid >> 5;
    const int mhalf = lane >> 4;          // WMMA lane-half (K select)
    const int lm    = lane & 15;
    // wave -> tiles: nt shared by both tiles so ONE set of B fragments serves both
    const int nt    = wave & 15;
    const int mtp   = wave >> 4;          // 0..1
    const int n     = nt * 16 + lm;       // hidden unit (output column)
    const int arow0 = mtp * 32 + lm;      // batch row, tile 0
    const int arow1 = arow0 + 16;         // batch row, tile 1

    // ---- one-time init: pack [Whh | Wcomb] -> f16 LDS, zero h-cols of abuf0 ----
    for (int idx = tid; idx < N_HID * N_HID; idx += 1024)
        wcat[(idx >> 8) * KPAD + (idx & 255)] = (_Float16)Whh[idx];
    for (int idx = tid; idx < N_HID * N_IN; idx += 1024)
        wcat[(idx >> 5) * KPAD + N_HID + (idx & 31)] = (_Float16)Wcomb[idx];
    for (int idx = tid; idx < BB * N_HID; idx += 1024)
        abuf0[(idx >> 8) * KPAD + (idx & 255)] = (_Float16)0.f;

    // ---- stage x_0 into abuf0 x-cols; prime async pipe with x_1 ----
#if USE_ASYNC_LDS
    const unsigned xs_base = (unsigned)(unsigned long long)xstg; // low 32 bits = LDS addr
    #pragma unroll
    for (int k = 0; k < 2; ++k) {                                // async x_0 -> xstg buf0
        const int idx = tid + k * 1024, b = idx >> 5, i = idx & 31;
        const unsigned long long gp =
            (unsigned long long)&input[(size_t)b * TT * N_IN + i];
        asm volatile("global_load_async_to_lds_b32 %0, %1, off"
                     :: "v"(xs_base + (unsigned)idx * 4u), "v"(gp) : "memory");
    }
    asm volatile("s_wait_asynccnt 0x0" ::: "memory");
    #pragma unroll
    for (int k = 0; k < 2; ++k) {                                // convert x_0 -> abuf0
        const int idx = tid + k * 1024, b = idx >> 5, i = idx & 31;
        abuf0[b * KPAD + N_HID + i] = (_Float16)xstg[idx];
    }
    #pragma unroll
    for (int k = 0; k < 2; ++k) {                                // async x_1 -> xstg buf1
        const int idx = tid + k * 1024, b = idx >> 5, i = idx & 31;
        const unsigned long long gp =
            (unsigned long long)&input[(size_t)b * TT * N_IN + N_IN + i];
        asm volatile("global_load_async_to_lds_b32 %0, %1, off"
                     :: "v"(xs_base + (unsigned)(BB * N_IN + idx) * 4u), "v"(gp) : "memory");
    }
#else
    for (int k = 0; k < 2; ++k) {
        const int idx = tid + k * 1024, b = idx >> 5, i = idx & 31;
        abuf0[b * KPAD + N_HID + i] = (_Float16)input[(size_t)b * TT * N_IN + i];
    }
#endif
    __syncthreads();

    // ---- hoist time-invariant B fragments into registers (9 chunks, K=288) ----
    v16h bfrag[9];
    #pragma unroll
    for (int kc = 0; kc < 9; ++kc) {
        const int kbB = kc * 32 + (mhalf ? 8 : 0);    // B 32x16: lanes 0-15 K{0..7,16..23}, 16-31 K{8..15,24..31}
        bfrag[kc] = load_frag16(&wcat[n * KPAD + kbB], &wcat[n * KPAD + kbB + 16]);
    }
    const float bias = bcat[n];

    // ---- persistent h state in registers ----
    float h0[8], h1[8];
    #pragma unroll
    for (int r = 0; r < 8; ++r) { h0[r] = 0.f; h1[r] = 0.f; }

    const int kb0 = mhalf ? 8 : 0;  // A 16x32: lanes 0-15 K{0..7,16..23}, 16-31 K{8..15,24..31}

    for (int t = 0; t < TT; ++t) {
        const _Float16* acur = (t & 1) ? abuf1 : abuf0;
        _Float16*       anxt = (t & 1) ? abuf0 : abuf1;
        const _Float16* a0p  = acur + arow0 * KPAD;
        const _Float16* a1p  = acur + arow1 * KPAD;

        // -- pipelined A loads + 18 WMMAs (2 tiles x 9 K-chunks, shared B) --
        v16h a0 = load_frag16(a0p + kb0, a0p + kb0 + 16);
        v16h a1 = load_frag16(a1p + kb0, a1p + kb0 + 16);
        v8f c0 = {}, c1 = {};
        #pragma unroll
        for (int kc = 0; kc < 9; ++kc) {
            v16h p0 = a0, p1 = a1;
            if (kc < 8) {                                  // prefetch next chunk's A
                const int kb = (kc + 1) * 32 + kb0;
                a0 = load_frag16(a0p + kb, a0p + kb + 16);
                a1 = load_frag16(a1p + kb, a1p + kb + 16);
            }
            c0 = __builtin_amdgcn_wmma_f32_16x16x32_f16(false, p0, false, bfrag[kc],
                                                        (short)0, c0, false, false);
            c1 = __builtin_amdgcn_wmma_f32_16x16x32_f16(false, p1, false, bfrag[kc],
                                                        (short)0, c1, false, false);
        }

        // -- leak/relu update in registers; write h(t+1) f16 into NEXT A buffer,
        //    store hidden[:,t,:] straight from registers (64B segments per row) --
        #pragma unroll
        for (int r = 0; r < 8; ++r) {
            const int m0 = mtp * 32 + mhalf * 8 + r;       // tile0 batch row
            const int m1 = m0 + 16;                        // tile1 batch row
            float pre0 = c0[r] + bias;
            float pre1 = c1[r] + bias;
            float a0f  = pre0 > 0.f ? pre0 : 0.f;
            float a1f  = pre1 > 0.f ? pre1 : 0.f;
            h0[r] = (1.f - ALPHA) * h0[r] + ALPHA * a0f;
            h1[r] = (1.f - ALPHA) * h1[r] + ALPHA * a1f;
            hidden[(size_t)m0 * TT * N_HID + (size_t)t * N_HID + n] = h0[r];
            hidden[(size_t)m1 * TT * N_HID + (size_t)t * N_HID + n] = h1[r];
            anxt[m0 * KPAD + n] = (_Float16)h0[r];
            anxt[m1 * KPAD + n] = (_Float16)h1[r];
        }

        // -- feed the x pipeline --
#if USE_ASYNC_LDS
        asm volatile("s_wait_asynccnt 0x0" ::: "memory");  // x_{t+1} landed in xstg
        if (t + 1 < TT) {
            #pragma unroll
            for (int k = 0; k < 2; ++k) {                  // convert x_{t+1} -> anxt
                const int idx = tid + k * 1024, b = idx >> 5, i = idx & 31;
                anxt[b * KPAD + N_HID + i] = (_Float16)xstg[((t + 1) & 1) * (BB * N_IN) + idx];
            }
        }
        if (t + 2 < TT) {
            #pragma unroll
            for (int k = 0; k < 2; ++k) {                  // async x_{t+2} -> xstg[t&1]
                const int idx = tid + k * 1024, b = idx >> 5, i = idx & 31;
                const unsigned long long gp = (unsigned long long)
                    &input[(size_t)b * TT * N_IN + (size_t)(t + 2) * N_IN + i];
                asm volatile("global_load_async_to_lds_b32 %0, %1, off"
                             :: "v"(xs_base + (unsigned)((t & 1) * (BB * N_IN) + idx) * 4u),
                                "v"(gp) : "memory");
            }
        }
#else
        if (t + 1 < TT) {
            #pragma unroll
            for (int k = 0; k < 2; ++k) {
                const int idx = tid + k * 1024, b = idx >> 5, i = idx & 31;
                const float* gp = &input[(size_t)b * TT * N_IN + (size_t)(t + 1) * N_IN + i];
                anxt[b * KPAD + N_HID + i] = (_Float16)(*gp);
                if (t + 2 < TT) __builtin_prefetch(gp + N_IN, 0, 0);
            }
        }
#endif
        __syncthreads();   // the single per-step barrier (double-buffered A)
    }
}

// ---------------------------------------------------------------------------
// Kernel 2: out[bt][o] = hidden[bt][:] . out_W[o][:] + out_b  (N padded 8->16)
// v4f loads + v_cvt_pk_rtz_f16_f32 packing feed f16 WMMA.
// ---------------------------------------------------------------------------
union F16x16 { v16h v; v2hf p[8]; };   // same 32-byte layout, different element views

__launch_bounds__(256, 4)
__global__ void outproj_kernel(const float* __restrict__ hidden, // [BT, H]
                               const float* __restrict__ out_W,  // [O, H]
                               const float* __restrict__ out_b,  // [O]
                               float* __restrict__ outp)         // [BT, O]
{
    const int lane  = threadIdx.x & 31;
    const int wave  = threadIdx.x >> 5;
    const int tile  = blockIdx.x * 8 + wave;    // BT/16 = 8192 tiles
    const int mhalf = lane >> 4;
    const int lm    = lane & 15;
    const size_t mrow = (size_t)tile * 16 + lm;

    v2hf zero2;
    zero2.x = (__fp16)0.f;
    zero2.y = (__fp16)0.f;

    v8f c = {};
    #pragma unroll
    for (int kc = 0; kc < 8; ++kc) {            // K = 256
        const int kbA = kc * 32 + (mhalf ? 8 : 0);
        F16x16 a;
        #pragma unroll
        for (int half = 0; half < 2; ++half) {
            v4f x0 = *(const v4f*)&hidden[mrow * N_HID + kbA + half * 16];
            v4f x1 = *(const v4f*)&hidden[mrow * N_HID + kbA + half * 16 + 4];
            a.p[half * 4 + 0] = __builtin_amdgcn_cvt_pkrtz(x0.x, x0.y);
            a.p[half * 4 + 1] = __builtin_amdgcn_cvt_pkrtz(x0.z, x0.w);
            a.p[half * 4 + 2] = __builtin_amdgcn_cvt_pkrtz(x1.x, x1.y);
            a.p[half * 4 + 3] = __builtin_amdgcn_cvt_pkrtz(x1.z, x1.w);
        }
        const int kbB = kc * 32 + (mhalf ? 8 : 0);
        F16x16 bm;
        #pragma unroll
        for (int j = 0; j < 8; ++j) bm.p[j] = zero2;
        if (lm < N_OUT) {                        // reconverges before WMMA (EXEC all-1s there)
            #pragma unroll
            for (int j = 0; j < 4; ++j) {
                v4f w = *(const v4f*)&out_W[lm * N_HID + kbB + (j >> 1) * 16 + (j & 1) * 4];
                bm.p[2 * j]     = __builtin_amdgcn_cvt_pkrtz(w.x, w.y);
                bm.p[2 * j + 1] = __builtin_amdgcn_cvt_pkrtz(w.z, w.w);
            }
        }
        c = __builtin_amdgcn_wmma_f32_16x16x32_f16(false, a.v, false, bm.v,
                                                   (short)0, c, false, false);
    }

    if (lm < N_OUT) {
        const float bias = out_b[lm];
        #pragma unroll
        for (int r = 0; r < 8; ++r) {
            const size_t m = (size_t)tile * 16 + mhalf * 8 + r;
            outp[m * N_OUT + lm] = c[r] + bias;
        }
    }
}

// ---------------------------------------------------------------------------
extern "C" void kernel_launch(void* const* d_in, const int* in_sizes, int n_in,
                              void* d_out, int out_size, void* d_ws, size_t ws_size,
                              hipStream_t stream) {
    const float* input = (const float*)d_in[0];
    const float* in_W  = (const float*)d_in[1];
    const float* in_b  = (const float*)d_in[2];
    const float* Wih   = (const float*)d_in[3];
    const float* bih   = (const float*)d_in[4];
    const float* Whh   = (const float*)d_in[5];
    const float* bhh   = (const float*)d_in[6];
    const float* out_W = (const float*)d_in[7];
    const float* out_b = (const float*)d_in[8];

    float* hidden = (float*)d_out;                      // [B,T,H]
    float* outp   = hidden + (size_t)BB * TT * N_HID;   // [B,T,O]
    float* Wcomb  = (float*)d_ws;                       // [H,I]
    float* bcat   = Wcomb + N_HID * N_IN;               // [H]

    prep_kernel<<<1, 256, 0, stream>>>(in_W, in_b, Wih, bih, bhh, Wcomb, bcat);

    const size_t smem = WCAT_BYTES + ABUF_BYTES + XSTG_BYTES;   // ~238 KB of 320 KB/WGP
    scan_kernel<<<1, 1024, smem, stream>>>(input, Whh, Wcomb, bcat, hidden);

    outproj_kernel<<<(BB * TT / 16) / 8, 256, 0, stream>>>(hidden, out_W, out_b, outp);
}
